// RNNModule_1391569403960
// MI455X (gfx1250) — matrix operand order repaired and checked by reading, not compiled, not run on hardware
//
#include <hip/hip_runtime.h>
#include <stdint.h>

// ---------------------------------------------------------------------------
// Types for CDNA5 WMMA (wave32): V_WMMA_F32_16X16X32_BF16
// ---------------------------------------------------------------------------
typedef __attribute__((ext_vector_type(16))) __bf16 v16bf;
typedef __attribute__((ext_vector_type(8)))  float  v8f;
typedef __attribute__((ext_vector_type(4)))  unsigned int u32x4;

union Frag { v16bf v; u32x4 q[2]; };

constexpr int TT  = 2048;          // sequence length
constexpr int BB  = 128;           // batch
constexpr int TB  = TT * BB;       // 262144 rows
constexpr int HH  = 64;            // hidden
constexpr int G4H = 256;           // 4*H gate width

// Branchless activations: raw v_exp_f32 / v_rcp_f32 / v_tanh_f32 (gfx1250).
__device__ __forceinline__ float fast_sigm(float x) {
  float t = __builtin_amdgcn_exp2f(x * -1.442695040888963f);   // v_exp_f32
  return __builtin_amdgcn_rcpf(1.0f + t);                      // v_rcp_f32
}
__device__ __forceinline__ float fast_tanh(float x) {
#if __has_builtin(__builtin_amdgcn_tanhf)
  return __builtin_amdgcn_tanhf(x);                            // v_tanh_f32
#else
  float t = __builtin_amdgcn_exp2f(x * 2.885390081777927f);    // exp(2x)
  return 1.0f - 2.0f * __builtin_amdgcn_rcpf(t + 1.0f);
#endif
}

// ---------------------------------------------------------------------------
// f32 -> bf16 conversion helpers
// ---------------------------------------------------------------------------
__global__ void cvt_bf16(const float* __restrict__ s, __bf16* __restrict__ d, int n) {
  int i = blockIdx.x * 256 + threadIdx.x;
  if (i < n) d[i] = (__bf16)s[i];
}

// x [B][T][32] f32  ->  xt [T][B][32] bf16   (time-major for the scan)
__global__ void transpose_x(const float* __restrict__ x, __bf16* __restrict__ xt) {
  int i = blockIdx.x * 256 + threadIdx.x;      // i = t*4096 + b*32 + k
  int k = i & 31;
  int b = (i >> 5) & 127;
  int t = i >> 12;
  if (t < TT) xt[i] = (__bf16)x[((size_t)b * TT + t) * 32 + k];
}

// ---------------------------------------------------------------------------
// Generic bf16 WMMA GEMM:  C[M][N] = act(A[M][K] * W[N][K]^T + bias)
// One wave computes a 16-row x 64-col strip (4 WMMA tiles, A reused 4x).
// A fragment (contiguous k order used here, same as B): lane<16 row=l, K [0..15];
// lane>=16 same row, K [16..31]  -> two 16B loads per K-step.
// B fragment: lane n holds col (n&15), K = (n>>4)*16 .. +15 contiguous (32B).
// C layout: lane holds col (l&15); VGPR j -> row j + 8*(l>>4).
// ---------------------------------------------------------------------------
template <bool OUT_BF16, bool RELU>
__global__ __launch_bounds__(256) void gemm_wmma(
    const __bf16* __restrict__ A, int lda,
    const __bf16* __restrict__ W,             // [N][K] row-major
    const float*  __restrict__ bias,          // [N] or nullptr
    void* __restrict__ Cout, int ldc,
    int M, int N, int K) {
  const int wave = threadIdx.x >> 5;
  const int lane = threadIdx.x & 31;
  const int l15  = lane & 15;
  const int lh   = lane >> 4;
  const int ngroups = N >> 6;                 // 64-col strips
  int flat  = blockIdx.x * 8 + wave;
  int mtile = flat / ngroups;
  int ng    = flat - mtile * ngroups;
  if (mtile * 16 >= M) return;

  const int row = mtile * 16 + l15;
  v8f acc[4] = {};
  const int kTiles = K >> 5;
  for (int kt = 0; kt < kTiles; ++kt) {
    Frag Af;
    const int koff = lh * 16 + kt * 32;
    Af.q[0] = *(const u32x4*)(A + (size_t)row * lda + koff);
    Af.q[1] = *(const u32x4*)(A + (size_t)row * lda + koff + 8);
    const int kb = lh * 16 + kt * 32;
#pragma unroll
    for (int g = 0; g < 4; ++g) {
      const int n = (ng * 4 + g) * 16 + l15;
      Frag Bf;
      const u32x4* pb = (const u32x4*)(W + (size_t)n * K + kb);
      Bf.q[0] = pb[0];
      Bf.q[1] = pb[1];
      acc[g] = __builtin_amdgcn_wmma_f32_16x16x32_bf16(
          false, Af.v, false, Bf.v, (short)0, acc[g], false, false);
    }
  }
  const int m0 = mtile * 16 + lh * 8;
#pragma unroll
  for (int g = 0; g < 4; ++g) {
    const int n = (ng * 4 + g) * 16 + l15;
    const float bv = bias ? bias[n] : 0.0f;
#pragma unroll
    for (int j = 0; j < 8; ++j) {
      float v = acc[g][j] + bv;
      if (RELU) v = fmaxf(v, 0.0f);
      if (OUT_BF16) ((__bf16*)Cout)[(size_t)(m0 + j) * ldc + n] = (__bf16)v;
      else          ((float*)Cout)[(size_t)(m0 + j) * ldc + n] = v;
    }
  }
}

// ---------------------------------------------------------------------------
// Persistent bidirectional LSTM scan, one block per direction.
// 256 threads = 8 waves (2 waves/SIMD -> big VGPR budget, no spills).
// wave w: q = w&3 (gate-column strip), mtiles {(w>>2)*4 .. +3} (batch rows).
// Wave owns tiles (mtile, {q,q+4,q+8,q+12}) = gates i,f,g,o for the same
// 16x16 (batch x hidden) region -> cell state c stays in registers (4 x v8f).
// w_hh fragments pinned in VGPRs (64) for all 2048 steps; h double-buffered
// in LDS with ONE barrier per step:
//   step s: read A from hbuf[(s+1)&1] (written at s-1), write h to hbuf[s&1].
// ---------------------------------------------------------------------------
__global__ __launch_bounds__(256) void lstm_scan(
    const float*  __restrict__ pre,   // [2][T][128][256] f32
    const __bf16* __restrict__ whh,   // [2][256][64]     bf16
    __bf16*       __restrict__ out)   // [T][128][128]    bf16 (fwd|bwd halves)
{
  const int dir  = blockIdx.x;
  const int wave = threadIdx.x >> 5;
  const int lane = threadIdx.x & 31;
  const int q     = wave & 3;
  const int mbase = (wave >> 2) * 4;    // first of 4 mtiles for this wave
  const int l15   = lane & 15;
  const int lh    = lane >> 4;

  __shared__ __bf16 hbuf[2][BB * HH];   // 2 x 16 KB double buffer

  pre += (size_t)dir * TT * BB * G4H;
  whh += (size_t)dir * G4H * HH;

  // Preload B fragments (w_hh^T tiles): [4 gates][2 K-steps], 64 VGPRs.
  Frag Bf[4][2];
#pragma unroll
  for (int g = 0; g < 4; ++g) {
    const int n = (q + g * 4) * 16 + l15;
#pragma unroll
    for (int kt = 0; kt < 2; ++kt) {
      const int kb = lh * 16 + kt * 32;
      const u32x4* p = (const u32x4*)(whh + (size_t)n * HH + kb);
      Bf[g][kt].q[0] = p[0];
      Bf[g][kt].q[1] = p[1];
    }
  }

  // Zero hbuf[1] (h_{-1} = 0; step 0 reads buffer 1).
  {
    unsigned* z = (unsigned*)&hbuf[1][0];
    for (int i = threadIdx.x; i < BB * HH / 2; i += 256) z[i] = 0u;
  }
  __syncthreads();

  v8f c_state[4] = {};                  // cell state per owned mtile
  const int hcol = q * 16 + l15;        // hidden column owned by this lane

  for (int s = 0; s < TT; ++s) {
    const int t = dir ? (TT - 1 - s) : s;
    const float* preT = pre + (size_t)t * BB * G4H;

    // Prefetch next step's pre slab (128 KB spread over 256 threads).
    if (s + 1 < TT) {
      const int tn = dir ? (TT - 2 - s) : (s + 1);
      __builtin_prefetch(pre + (size_t)tn * BB * G4H + threadIdx.x * 128, 0, 0);
    }

    const __bf16* hr = &hbuf[(s + 1) & 1][0];   // h from previous step
    __bf16*       hw = &hbuf[s & 1][0];         // h produced this step

#pragma unroll
    for (int mi = 0; mi < 4; ++mi) {
      const int mtile = mbase + mi;
      const int arow  = mtile * 16 + l15;
      const int m0    = mtile * 16 + lh * 8;

      // A fragments (h_{t-1}) from LDS: two 16B chunks per K-step.
      Frag Af[2];
#pragma unroll
      for (int kt = 0; kt < 2; ++kt) {
        const int koff = lh * 16 + kt * 32;
        Af[kt].q[0] = *(const u32x4*)(hr + (size_t)arow * HH + koff);
        Af[kt].q[1] = *(const u32x4*)(hr + (size_t)arow * HH + koff + 8);
      }

      // z = pre[t] + h @ w_hh^T   (accumulators seeded from pre)
      v8f acc[4];
#pragma unroll
      for (int g = 0; g < 4; ++g) {
        const int col = (q + g * 4) * 16 + l15;
        const float* cp = preT + (size_t)m0 * G4H + col;
        v8f ci;
#pragma unroll
        for (int j = 0; j < 8; ++j) ci[j] = cp[(size_t)j * G4H];
        ci = __builtin_amdgcn_wmma_f32_16x16x32_bf16(false, Af[0].v, false, Bf[g][0].v,
                                                     (short)0, ci, false, false);
        ci = __builtin_amdgcn_wmma_f32_16x16x32_bf16(false, Af[1].v, false, Bf[g][1].v,
                                                     (short)0, ci, false, false);
        acc[g] = ci;
      }

      // Gates (PyTorch order i,f,g,o): branchless v_exp/v_rcp/v_tanh.
      __bf16* op = out + ((size_t)t * BB + m0) * (2 * HH) + dir * HH + hcol;
#pragma unroll
      for (int j = 0; j < 8; ++j) {
        const float zi = acc[0][j], zf = acc[1][j], zg = acc[2][j], zo = acc[3][j];
        const float cs = fast_sigm(zf) * c_state[mi][j] + fast_sigm(zi) * fast_tanh(zg);
        c_state[mi][j] = cs;
        const float hv = fast_sigm(zo) * fast_tanh(cs);
        const __bf16 hb = (__bf16)hv;
        hw[(size_t)(m0 + j) * HH + hcol] = hb;   // LDS for next step's A
        op[(size_t)j * (2 * HH)] = hb;           // layer output (global)
      }
    }

    __syncthreads();   // h complete in LDS; orders reads vs next step's writes
  }
}

// ---------------------------------------------------------------------------
// Attention head
// ---------------------------------------------------------------------------
__global__ void attn_scores(const __bf16* __restrict__ hs,   // [T][B][128]
                            const float* __restrict__ w_attn,
                            const float* __restrict__ b_attn,
                            float* __restrict__ scores) {    // [B][T]
  int i = blockIdx.x * 256 + threadIdx.x;
  int b = i & 127, t = i >> 7;
  if (t >= TT) return;
  const __bf16* p = hs + ((size_t)t * BB + b) * (2 * HH);
  float s = b_attn[0];
#pragma unroll 4
  for (int h = 0; h < 2 * HH; ++h) s += (float)p[h] * w_attn[h];
  scores[(size_t)b * TT + t] = s;
}

__global__ __launch_bounds__(256) void attn_pool(const float* __restrict__ scores,
                                                 const __bf16* __restrict__ hs,
                                                 float* __restrict__ pooled) { // [B][128]
  __shared__ float sc[TT];
  __shared__ float red[8];
  __shared__ float pr[256];
  const int b = blockIdx.x, tid = threadIdx.x;

  float mx = -1e30f;
  for (int t = tid; t < TT; t += 256) {
    float v = scores[(size_t)b * TT + t];
    sc[t] = v;
    mx = fmaxf(mx, v);
  }
  for (int o = 16; o > 0; o >>= 1) mx = fmaxf(mx, __shfl_xor(mx, o, 32));
  if ((tid & 31) == 0) red[tid >> 5] = mx;
  __syncthreads();
  float bm = red[0];
#pragma unroll
  for (int w = 1; w < 8; ++w) bm = fmaxf(bm, red[w]);
  __syncthreads();

  float sum = 0.0f;
  for (int t = tid; t < TT; t += 256) {
    float e = __builtin_amdgcn_exp2f((sc[t] - bm) * 1.442695040888963f);
    sc[t] = e;
    sum += e;
  }
  for (int o = 16; o > 0; o >>= 1) sum += __shfl_xor(sum, o, 32);
  if ((tid & 31) == 0) red[tid >> 5] = sum;
  __syncthreads();
  float ts = 0.0f;
#pragma unroll
  for (int w = 0; w < 8; ++w) ts += red[w];
  const float inv = 1.0f / ts;

  const int h = tid & 127, half = tid >> 7;
  float acc = 0.0f;
  for (int t = half * (TT / 2); t < (half + 1) * (TT / 2); ++t)
    acc += sc[t] * (float)hs[((size_t)t * BB + b) * (2 * HH) + h];
  pr[tid] = acc;
  __syncthreads();
  if (tid < 128) pooled[(size_t)b * 128 + tid] = (pr[tid] + pr[tid + 128]) * inv;
}

// BN1 -> fc1+relu -> fc2+elu -> BN2 -> fc3 ; one block per batch row.
__global__ __launch_bounds__(128) void head_mlp(
    const float* __restrict__ pooled,
    const float* g1, const float* be1, const float* m1, const float* v1,
    const float* w1, const float* b1, const float* w2, const float* b2,
    const float* g2, const float* be2, const float* m2, const float* v2,
    const float* w3, const float* b3, float* __restrict__ outy) {
  const int b = blockIdx.x, n = threadIdx.x;
  __shared__ float y0[128], y1[128], y2[64];
  const float x = pooled[(size_t)b * 128 + n];
  y0[n] = (x - m1[n]) * rsqrtf(v1[n] + 1e-5f) * g1[n] + be1[n];
  __syncthreads();
  float a = b1[n];
#pragma unroll 4
  for (int k = 0; k < 128; ++k) a += y0[k] * w1[(size_t)n * 128 + k];
  y1[n] = fmaxf(a, 0.0f);
  __syncthreads();
  if (n < 64) {
    float a2 = b2[n];
#pragma unroll 4
    for (int k = 0; k < 128; ++k) a2 += y1[k] * w2[(size_t)n * 128 + k];
    a2 = (a2 > 0.0f) ? a2 : (expf(a2) - 1.0f);   // ELU (exact tail)
    y2[n] = (a2 - m2[n]) * rsqrtf(v2[n] + 1e-5f) * g2[n] + be2[n];
  }
  __syncthreads();
  if (n < 3) {
    float a3 = b3[n];
#pragma unroll
    for (int k = 0; k < 64; ++k) a3 += y2[k] * w3[(size_t)n * 64 + k];
    outy[(size_t)b * 3 + n] = a3;
  }
}

// ---------------------------------------------------------------------------
// Host-side orchestration
// ---------------------------------------------------------------------------
extern "C" void kernel_launch(void* const* d_in, const int* in_sizes, int n_in,
                              void* d_out, int out_size, void* d_ws, size_t ws_size,
                              hipStream_t stream) {
  const float* x       = (const float*)d_in[0];
  const float* w_proj  = (const float*)d_in[1];
  const float* b_proj  = (const float*)d_in[2];
  const float* w_ih_l0 = (const float*)d_in[3];
  const float* w_hh_l0 = (const float*)d_in[4];
  const float* b_l0    = (const float*)d_in[5];
  const float* w_ih_r  = (const float*)d_in[6];
  const float* w_hh_r  = (const float*)d_in[7];
  const float* b_r     = (const float*)d_in[8];
  const float* w_attn  = (const float*)d_in[9];
  const float* b_attn  = (const float*)d_in[10];
  const float* g1  = (const float*)d_in[11];
  const float* be1 = (const float*)d_in[12];
  const float* m1  = (const float*)d_in[13];
  const float* v1  = (const float*)d_in[14];
  const float* w1  = (const float*)d_in[15];
  const float* b1  = (const float*)d_in[16];
  const float* w2  = (const float*)d_in[17];
  const float* b2  = (const float*)d_in[18];
  const float* g2  = (const float*)d_in[19];
  const float* be2 = (const float*)d_in[20];
  const float* m2  = (const float*)d_in[21];
  const float* v2  = (const float*)d_in[22];
  const float* w3  = (const float*)d_in[23];
  const float* b3  = (const float*)d_in[24];

  char* ws = (char*)d_ws;
  size_t off = 0;
  auto alloc = [&](size_t bytes) -> void* {
    void* p = ws + off;
    off += (bytes + 255) & ~(size_t)255;
    return p;
  };

  __bf16* xt     = (__bf16*)alloc((size_t)TB * 32 * 2);          // 16 MB
  __bf16* bufA   = (__bf16*)alloc((size_t)TB * 128 * 2);         // 64 MB
  __bf16* bufB   = (__bf16*)alloc((size_t)TB * 128 * 2);         // 64 MB
  float*  preBuf = (float*) alloc((size_t)2 * TB * G4H * 4);     // 512 MB
  __bf16* wp_bf    = (__bf16*)alloc((size_t)64 * 32 * 2);
  __bf16* wihl0_bf = (__bf16*)alloc((size_t)2 * G4H * 64 * 2);
  __bf16* whhl0_bf = (__bf16*)alloc((size_t)2 * G4H * 64 * 2);
  __bf16* wihr_bf  = (__bf16*)alloc((size_t)3 * 2 * G4H * 128 * 2);
  __bf16* whhr_bf  = (__bf16*)alloc((size_t)3 * 2 * G4H * 64 * 2);
  float*  scores   = (float*)alloc((size_t)BB * TT * 4);
  float*  pooled   = (float*)alloc((size_t)BB * 128 * 4);

  auto cvt = [&](const float* s, __bf16* d, int n) {
    cvt_bf16<<<(n + 255) / 256, 256, 0, stream>>>(s, d, n);
  };
  cvt(w_proj,  wp_bf,    64 * 32);
  cvt(w_ih_l0, wihl0_bf, 2 * G4H * 64);
  cvt(w_hh_l0, whhl0_bf, 2 * G4H * 64);
  cvt(w_ih_r,  wihr_bf,  3 * 2 * G4H * 128);
  cvt(w_hh_r,  whhr_bf,  3 * 2 * G4H * 64);

  transpose_x<<<(TB * 32) / 256, 256, 0, stream>>>(x, xt);

  // Input projection + ReLU: [TB,32] x [64,32]^T -> bufA [TB,64] bf16
  gemm_wmma<true, true><<<(TB / 16) / 8, 256, 0, stream>>>(
      xt, 32, wp_bf, b_proj, bufA, 64, TB, 64, 32);

  __bf16* inBuf = bufA;
  __bf16* outBuf = bufB;
  int din = 64;
  for (int l = 0; l < 4; ++l) {
    const __bf16* wih = (l == 0) ? wihl0_bf : wihr_bf + (size_t)(l - 1) * 2 * G4H * 128;
    const __bf16* whh = (l == 0) ? whhl0_bf : whhr_bf + (size_t)(l - 1) * 2 * G4H * 64;
    const float*  bia = (l == 0) ? b_l0 : b_r + (size_t)(l - 1) * 2 * G4H;
    for (int d = 0; d < 2; ++d) {
      // pre[d] = inBuf @ w_ih[d]^T + b[d]   : [TB, 256] f32
      gemm_wmma<false, false><<<((TB / 16) * (G4H / 64)) / 8, 256, 0, stream>>>(
          inBuf, din, wih + (size_t)d * G4H * din, bia + (size_t)d * G4H,
          preBuf + (size_t)d * TB * G4H, G4H, TB, G4H, din);
    }
    lstm_scan<<<2, 256, 0, stream>>>(preBuf, whh, outBuf);
    inBuf = outBuf;
    outBuf = (inBuf == bufA) ? bufB : bufA;
    din = 128;
  }

  attn_scores<<<(BB * TT) / 256, 256, 0, stream>>>(inBuf, w_attn, b_attn, scores);
  attn_pool<<<BB, 256, 0, stream>>>(scores, inBuf, pooled);
  head_mlp<<<BB, 128, 0, stream>>>(pooled, g1, be1, m1, v1, w1, b1, w2, b2,
                                   g2, be2, m2, v2, w3, b3, (float*)d_out);
}
